// IPAttnProcessor_5961414607543
// MI455X (gfx1250) — matrix operand order repaired and checked by reading, compiled, not run on hardware
//
#include <hip/hip_runtime.h>
#include <hip/hip_bf16.h>
#include <math.h>

typedef __bf16 bf16;
typedef __attribute__((ext_vector_type(16))) __bf16 v16bf;
typedef __attribute__((ext_vector_type(8)))  float  v8f;
typedef __attribute__((ext_vector_type(4)))  float  f32x4;

union FragAB { f32x4 f4[2]; v16bf v; };   // 32B: one WMMA A or B operand
union Chunk8 { f32x4 f4; bf16 h[8]; };    // 16B: 8 bf16

__device__ __forceinline__ v8f zero8() {
  v8f z;
  for (int i = 0; i < 8; ++i) z[i] = 0.f;
  return z;
}

// ---------------------------------------------------------------------------
// f32 -> bf16 conversion (memory-bound, one-time)
// ---------------------------------------------------------------------------
__global__ void cvt_f32_bf16(const float* __restrict__ in, bf16* __restrict__ out, int n) {
  for (int i = blockIdx.x * blockDim.x + threadIdx.x; i < n; i += gridDim.x * blockDim.x)
    out[i] = (bf16)in[i];
}

// encoder (8,81,768) -> ctx (8,77,768) bf16 + ip (8,4,768) bf16
__global__ void split_enc(const float* __restrict__ enc, bf16* __restrict__ ctx,
                          bf16* __restrict__ ip) {
  const int n = 8 * 81 * 768;
  for (int i = blockIdx.x * blockDim.x + threadIdx.x; i < n; i += gridDim.x * blockDim.x) {
    int c = i % 768, rem = i / 768;
    int l = rem % 81, b = rem / 81;
    bf16 val = (bf16)enc[i];
    if (l < 77) ctx[(b * 77 + l) * 768 + c] = val;
    else        ip[(b * 4 + (l - 77)) * 768 + c] = val;
  }
}

// ---------------------------------------------------------------------------
// bf16 GEMM: C[M,N] = A[M,K] @ B[K,N] (+bias), f32 accumulate via WMMA
// block tile 128x128, K-step 64, double-buffered LDS, 8 waves (wave32)
// ---------------------------------------------------------------------------
#define GT_M 128
#define GT_N 128
#define GT_K 64
#define A_PITCH 36                     // dwords per A LDS row (32 data + 4 pad) -> conflict-free b128
#define B_PITCH 36                     // dwords per Bt LDS row (Bt = B transposed, k contiguous)
#define A_STAGE (GT_M * A_PITCH)
#define B_STAGE (GT_N * B_PITCH)
#define GEMM_LDS_BYTES ((2 * A_STAGE + 2 * B_STAGE) * 4)   // 72 KB

__global__ __launch_bounds__(256) void gemm_bf16(
    const bf16* __restrict__ A, const bf16* __restrict__ B,
    bf16* __restrict__ Cb, float* __restrict__ Cf, const float* __restrict__ bias,
    int M, int N, int K)
{
  extern __shared__ uint lds[];
  uint* aL = lds;                   // 2 stages of A (row-major, m x k)
  uint* bL = lds + 2 * A_STAGE;     // 2 stages of Bt (n x k, k contiguous)

  const int tid  = threadIdx.x;
  const int lane = tid & 31, wave = tid >> 5;
  const int l16  = lane & 15, half = lane >> 4;
  const int wm = wave & 3, wn = wave >> 2;          // 4x2 wave grid, wave tile 32x64
  const long blockM = (long)blockIdx.y * GT_M;
  const long blockN = (long)blockIdx.x * GT_N;
  const int  KT = K / GT_K;

  v8f acc[2][4];
  for (int i = 0; i < 2; ++i)
    for (int j = 0; j < 4; ++j) acc[i][j] = zero8();

  f32x4  aReg[4];
  Chunk8 bReg[4];

  auto gload = [&](int kt) {                         // global -> regs (128x64 A, 64x128 B)
#pragma unroll
    for (int j = 0; j < 4; ++j) {
      int idx = tid + j * 256;
      {
        int row = idx >> 3, c8 = idx & 7;            // 8-bf16 chunks along K
        long r = blockM + row;
        if (r < M)
          aReg[j] = *(const f32x4*)(A + r * (long)K + (long)kt * GT_K + c8 * 8);
        else
          for (int e = 0; e < 4; ++e) aReg[j][e] = 0.f;
      }
      {
        int k = idx >> 4, n8 = idx & 15;             // 8-bf16 chunks along N
        bReg[j].f4 = *(const f32x4*)(B + (long)(kt * GT_K + k) * N + blockN + n8 * 8);
      }
    }
  };

  auto sstore = [&](int s) {                         // regs -> LDS (B transposed)
    uint* aS = aL + s * A_STAGE;
    bf16* bS = (bf16*)(bL + s * B_STAGE);
#pragma unroll
    for (int j = 0; j < 4; ++j) {
      int idx = tid + j * 256;
      {
        int row = idx >> 3, c8 = idx & 7;
        *(f32x4*)(aS + row * A_PITCH + c8 * 4) = aReg[j];
      }
      {
        int k = idx >> 4, n8 = idx & 15;
#pragma unroll
        for (int e = 0; e < 8; ++e)
          bS[(n8 * 8 + e) * (B_PITCH * 2) + k] = bReg[j].h[e];
      }
    }
  };

  gload(0);
  sstore(0);

  for (int kt = 0; kt < KT; ++kt) {
    __syncthreads();
    const bool more = (kt + 1) < KT;
    if (more) gload(kt + 1);

    const uint* aS = aL + (kt & 1) * A_STAGE;
    const uint* bS = bL + (kt & 1) * B_STAGE;
#pragma unroll
    for (int koi = 0; koi < 2; ++koi) {              // two 32-deep WMMA steps per stage
      const int ko2 = koi * 16;                      // ko/2 in dwords
      FragAB af[2];
#pragma unroll
      for (int mf = 0; mf < 2; ++mf) {
        const int m = wm * 32 + mf * 16 + l16;
        const uint* p = aS + m * A_PITCH + ko2 + half * 4;   // k = ko + half*8 + 0..7
        af[mf].f4[0] = *(const f32x4*)p;
        af[mf].f4[1] = *(const f32x4*)(p + 8);               // k = ko + 16 + half*8 + 0..7
      }
#pragma unroll
      for (int nf = 0; nf < 4; ++nf) {
        const int n = wn * 64 + nf * 16 + l16;
        const uint* p = bS + n * B_PITCH + ko2 + half * 4;   // k = ko + half*8 + 0..7
        FragAB bb;
        bb.f4[0] = *(const f32x4*)p;
        bb.f4[1] = *(const f32x4*)(p + 8);                   // k = ko + 16 + half*8 + 0..7
#pragma unroll
        for (int mf = 0; mf < 2; ++mf)
          acc[mf][nf] = __builtin_amdgcn_wmma_f32_16x16x32_bf16(
              false, af[mf].v, false, bb.v, (short)0, acc[mf][nf], false, false);
      }
    }
    if (more) sstore((kt + 1) & 1);
  }

  // epilogue: C layout row = r + 8*half, col = l16 within each 16x16 frag
#pragma unroll
  for (int mf = 0; mf < 2; ++mf)
#pragma unroll
    for (int r = 0; r < 8; ++r) {
      const long row = blockM + wm * 32 + mf * 16 + r + 8 * half;
      if (row >= M) continue;
#pragma unroll
      for (int nf = 0; nf < 4; ++nf) {
        const long col = blockN + wn * 64 + nf * 16 + l16;
        const float val = acc[mf][nf][r];
        if (Cf) Cf[row * N + col] = val + bias[col];
        else    Cb[row * N + col] = (bf16)val;
      }
    }
}

// ---------------------------------------------------------------------------
// Fused IP-adapter attention: per (b,h,128-query tile)
// 77 main keys + 4 IP keys packed into one 96-key padded tile:
//   keys 0..76 = main, 77..79 = zero pad, 80..83 = IP, 84..95 = zero pad
// Two disjoint softmaxes normalize disjoint key ranges of one probs tile, so a
// single probs@V WMMA pass yields main_attn + IP_SCALE * ip_attn.
// ---------------------------------------------------------------------------
#define DH 160
#define NKEY 96
#define KC_PITCH 168      // bf16 per Kc row (160 + 8 pad)    -> conflict-free b128
#define VT_PITCH 104      // bf16 per Vt row (96 keys + pad)  -> conflict-free b128
#define PR_PITCH 104
#define ATTN_LDS_BYTES (NKEY * KC_PITCH * 2 + DH * VT_PITCH * 2 + 8 * 16 * PR_PITCH * 2)  // 90 KB

__global__ __launch_bounds__(256) void ip_attn(
    const bf16* __restrict__ q, const bf16* __restrict__ k, const bf16* __restrict__ v,
    const bf16* __restrict__ kip, const bf16* __restrict__ vip, bf16* __restrict__ out)
{
  extern __shared__ char smem[];
  bf16* Kc = (bf16*)smem;                                   // [96][168]  (key, d)
  bf16* Vt = (bf16*)(smem + NKEY * KC_PITCH * 2);           // [160][104] (d, key) transposed
  bf16* Pr = (bf16*)(smem + NKEY * KC_PITCH * 2 + DH * VT_PITCH * 2);  // per-wave [16][104]

  const int bh = blockIdx.x;
  const int b = bh >> 3, h = bh & 7;
  const int tid = threadIdx.x, lane = tid & 31, wave = tid >> 5;
  const int l16 = lane & 15, half = lane >> 4;
  const long qbase = (long)blockIdx.y * 128 + wave * 16;
  const float sm_scale = 0.0790569415f;                     // 1/sqrt(160)

  // cooperative K/V staging (pad rows zeroed)
  for (int i = tid; i < NKEY * DH; i += 256) {
    const int key = i / DH, d = i % DH;
    float kv = 0.f, vv = 0.f;
    if (key < 77) {
      const long off = ((long)(b * 77 + key)) * 1280 + h * DH + d;
      kv = (float)k[off]; vv = (float)v[off];
    } else if (key >= 80 && key < 84) {
      const long off = ((long)(b * 4 + (key - 80))) * 1280 + h * DH + d;
      kv = (float)kip[off]; vv = (float)vip[off];
    }
    Kc[key * KC_PITCH + d] = (bf16)kv;
    Vt[d * VT_PITCH + key] = (bf16)vv;
  }
  __syncthreads();

  // ---- scores 16x96 = Q(16x160) @ K^T, GEMM-K = dh ----
  v8f s[6];
  for (int i = 0; i < 6; ++i) s[i] = zero8();
  const bf16* qrow = q + ((long)b * 4096 + qbase + l16) * 1280 + h * DH;
#pragma unroll
  for (int k5 = 0; k5 < 5; ++k5) {
    const int ko = k5 * 32;
    FragAB a;
    a.f4[0] = *(const f32x4*)(qrow + ko + half * 8);
    a.f4[1] = *(const f32x4*)(qrow + ko + 16 + half * 8);
#pragma unroll
    for (int nf = 0; nf < 6; ++nf) {
      const bf16* kp = Kc + (nf * 16 + l16) * KC_PITCH + ko + half * 8;
      FragAB bb;
      bb.f4[0] = *(const f32x4*)kp;
      bb.f4[1] = *(const f32x4*)(kp + 16);
      s[nf] = __builtin_amdgcn_wmma_f32_16x16x32_bf16(
          false, a.v, false, bb.v, (short)0, s[nf], false, false);
    }
  }

  // ---- dual softmax: main over keys<77, IP over keys 80..83 ----
  bf16* prow = Pr + wave * 16 * PR_PITCH;
#pragma unroll
  for (int r = 0; r < 8; ++r) {
    const int m = r + 8 * half;
    float mx = -1e30f;
#pragma unroll
    for (int nf = 0; nf < 5; ++nf) {
      const float sv = s[nf][r] * sm_scale;
      s[nf][r] = sv;
      if (nf * 16 + l16 < 77) mx = fmaxf(mx, sv);
    }
    const float s5 = s[5][r] * sm_scale;
    float ix = (l16 < 4) ? s5 : -1e30f;
#pragma unroll
    for (int off = 1; off < 16; off <<= 1) {           // reduce within 16-lane half
      mx = fmaxf(mx, __shfl_xor(mx, off, 32));
      ix = fmaxf(ix, __shfl_xor(ix, off, 32));
    }
    float p[6], sm = 0.f;
#pragma unroll
    for (int nf = 0; nf < 5; ++nf) {
      p[nf] = (nf * 16 + l16 < 77) ? __expf(s[nf][r] - mx) : 0.f;
      sm += p[nf];
    }
    p[5] = (l16 < 4) ? __expf(s5 - ix) : 0.f;
    float si = p[5];
#pragma unroll
    for (int off = 1; off < 16; off <<= 1) {
      sm += __shfl_xor(sm, off, 32);
      si += __shfl_xor(si, off, 32);
    }
    const float rm = 1.f / sm;
    const float ri = 1.0f /* IP_SCALE */ / si;
#pragma unroll
    for (int nf = 0; nf < 5; ++nf)
      prow[m * PR_PITCH + nf * 16 + l16] = (bf16)(p[nf] * rm);
    prow[m * PR_PITCH + 80 + l16] = (bf16)(p[5] * ri);   // lanes>=4 store exact 0 pad
  }
  // per-wave probs region: same-wave LDS RAW, DS ops are in-order -> no barrier

  // ---- out 16x160 = P(16x96) @ V(96x160), GEMM-K = key ----
  v8f o[10];
  for (int i = 0; i < 10; ++i) o[i] = zero8();
#pragma unroll
  for (int k3 = 0; k3 < 3; ++k3) {
    const int ko = k3 * 32;
    FragAB a;
    const bf16* pp = prow + l16 * PR_PITCH + ko;
    a.f4[0] = *(const f32x4*)(pp + half * 8);
    a.f4[1] = *(const f32x4*)(pp + 16 + half * 8);
#pragma unroll
    for (int nf = 0; nf < 10; ++nf) {
      const bf16* vp = Vt + (nf * 16 + l16) * VT_PITCH + ko + half * 8;
      FragAB bb;
      bb.f4[0] = *(const f32x4*)vp;
      bb.f4[1] = *(const f32x4*)(vp + 16);
      o[nf] = __builtin_amdgcn_wmma_f32_16x16x32_bf16(
          false, a.v, false, bb.v, (short)0, o[nf], false, false);
    }
  }

  bf16* orow = out + ((long)b * 4096 + qbase) * 1280 + h * DH;
#pragma unroll
  for (int r = 0; r < 8; ++r) {
    const int m = r + 8 * half;
#pragma unroll
    for (int nf = 0; nf < 10; ++nf)
      orow[(long)m * 1280 + nf * 16 + l16] = (bf16)o[nf][r];
  }
}

// ---------------------------------------------------------------------------
extern "C" void kernel_launch(void* const* d_in, const int* in_sizes, int n_in,
                              void* d_out, int out_size, void* d_ws, size_t ws_size,
                              hipStream_t stream)
{
  (void)in_sizes; (void)n_in; (void)out_size; (void)ws_size;
  const float* hs   = (const float*)d_in[0];
  const float* enc  = (const float*)d_in[1];
  const float* Wq   = (const float*)d_in[2];
  const float* Wk   = (const float*)d_in[3];
  const float* Wv   = (const float*)d_in[4];
  const float* Wkip = (const float*)d_in[5];
  const float* Wvip = (const float*)d_in[6];
  const float* Wo   = (const float*)d_in[7];
  const float* bo   = (const float*)d_in[8];
  float* out = (float*)d_out;

  const int B = 8, S = 4096, D = 1280, C = 768, H = 8;
  const long NS = (long)B * S;  // 32768

  char* ws = (char*)d_ws;
  size_t off = 0;
  auto carve = [&](size_t bytes) -> char* {
    char* p = ws + off;
    off += (bytes + 255) & ~(size_t)255;
    return p;
  };
  bf16* hs_b   = (bf16*)carve(NS * D * 2);
  bf16* Wq_b   = (bf16*)carve((size_t)D * D * 2);
  bf16* Wk_b   = (bf16*)carve((size_t)C * D * 2);
  bf16* Wv_b   = (bf16*)carve((size_t)C * D * 2);
  bf16* Wkip_b = (bf16*)carve((size_t)C * D * 2);
  bf16* Wvip_b = (bf16*)carve((size_t)C * D * 2);
  bf16* Wo_b   = (bf16*)carve((size_t)D * D * 2);
  bf16* ctx_b  = (bf16*)carve((size_t)B * 77 * C * 2);
  bf16* ip_b   = (bf16*)carve((size_t)B * 4 * C * 2);
  bf16* q_b    = (bf16*)carve(NS * D * 2);
  bf16* k_b    = (bf16*)carve((size_t)B * 77 * D * 2);
  bf16* v_b    = (bf16*)carve((size_t)B * 77 * D * 2);
  bf16* kip_b  = (bf16*)carve((size_t)B * 4 * D * 2);
  bf16* vip_b  = (bf16*)carve((size_t)B * 4 * D * 2);
  bf16* attn_b = (bf16*)carve(NS * D * 2);

  hipFuncSetAttribute((const void*)gemm_bf16, hipFuncAttributeMaxDynamicSharedMemorySize, GEMM_LDS_BYTES);
  hipFuncSetAttribute((const void*)ip_attn,   hipFuncAttributeMaxDynamicSharedMemorySize, ATTN_LDS_BYTES);

  const int T = 256;
  cvt_f32_bf16<<<2048, T, 0, stream>>>(hs, hs_b, (int)(NS * D));
  cvt_f32_bf16<<<512, T, 0, stream>>>(Wq, Wq_b, D * D);
  cvt_f32_bf16<<<512, T, 0, stream>>>(Wk, Wk_b, C * D);
  cvt_f32_bf16<<<512, T, 0, stream>>>(Wv, Wv_b, C * D);
  cvt_f32_bf16<<<512, T, 0, stream>>>(Wkip, Wkip_b, C * D);
  cvt_f32_bf16<<<512, T, 0, stream>>>(Wvip, Wvip_b, C * D);
  cvt_f32_bf16<<<512, T, 0, stream>>>(Wo, Wo_b, D * D);
  split_enc<<<512, T, 0, stream>>>(enc, ctx_b, ip_b);

  // q = hidden @ Wq  (32768x1280x1280)
  gemm_bf16<<<dim3(D / 128, (int)(NS / 128)), T, GEMM_LDS_BYTES, stream>>>(
      hs_b, Wq_b, q_b, nullptr, nullptr, (int)NS, D, D);
  // k/v projections (616x1280x768), IP (32x1280x768)
  gemm_bf16<<<dim3(D / 128, 5), T, GEMM_LDS_BYTES, stream>>>(ctx_b, Wk_b,  k_b,  nullptr, nullptr, 616, D, C);
  gemm_bf16<<<dim3(D / 128, 5), T, GEMM_LDS_BYTES, stream>>>(ctx_b, Wv_b,  v_b,  nullptr, nullptr, 616, D, C);
  gemm_bf16<<<dim3(D / 128, 1), T, GEMM_LDS_BYTES, stream>>>(ip_b,  Wkip_b, kip_b, nullptr, nullptr, 32, D, C);
  gemm_bf16<<<dim3(D / 128, 1), T, GEMM_LDS_BYTES, stream>>>(ip_b,  Wvip_b, vip_b, nullptr, nullptr, 32, D, C);
  // fused attention (main + IP in one WMMA pipeline)
  ip_attn<<<dim3(B * H, S / 128), T, ATTN_LDS_BYTES, stream>>>(q_b, k_b, v_b, kip_b, vip_b, attn_b);
  // out = attn @ Wo + bo, f32 result straight to d_out
  gemm_bf16<<<dim3(D / 128, (int)(NS / 128)), T, GEMM_LDS_BYTES, stream>>>(
      attn_b, Wo_b, nullptr, out, bo, (int)NS, D, D);
}
